// Attention_29841432772729
// MI455X (gfx1250) — matrix operand order repaired and checked by reading, not compiled, not run on hardware
//
#include <hip/hip_runtime.h>

typedef __attribute__((ext_vector_type(16))) __bf16 v16bf;
typedef __attribute__((ext_vector_type(8)))  float  v8f;
typedef __attribute__((ext_vector_type(2)))  float  v2f;
typedef __attribute__((ext_vector_type(2)))  __bf16 v2bf;

constexpr int kB = 2, kH = 16, kS = 2048, kD = 64;
constexpr int BLOCK_M = 128;   // Q rows per workgroup (8 waves x 16)
constexpr int BLOCK_N = 64;    // keys per KV tile
constexpr int NWAVE   = 8;
constexpr int LSTR    = 68;    // LDS row stride in bf16 elems (34 dwords, conflict-free)

// Native bf16 conversions: backend selects v_cvt_pk_bf16_f32 (RNE).
__device__ __forceinline__ unsigned bf16_2(float lo, float hi) {
  v2f x = {lo, hi};
  return __builtin_bit_cast(unsigned, __builtin_convertvector(x, v2bf));
}

// Raw v_exp_f32 (no denormal-range fixup). Inputs here are either in
// [-inf, 0] with moderate magnitude (score - rowmax) or ~-1e30 (masked),
// where hardware underflow-to-zero is exactly what we want.
__device__ __forceinline__ float fast_exp2(float x) {
#if __has_builtin(__builtin_amdgcn_exp2f)
  return __builtin_amdgcn_exp2f(x);
#else
  return exp2f(x);
#endif
}

union AFrag { v16bf v; unsigned u[8]; };

__global__ __launch_bounds__(256, 1) void flash_attn_fwd(
    const float* __restrict__ q, const float* __restrict__ k,
    const float* __restrict__ v, float* __restrict__ out) {
  __shared__ unsigned short k_lds[BLOCK_N * LSTR];          // K tile [key][d] bf16
  __shared__ unsigned short vt_lds[kD * LSTR];              // V tile transposed [d][key]
  __shared__ unsigned short p_lds[NWAVE * 16 * LSTR];       // per-wave P staging

  const int tid  = threadIdx.x;
  const int lane = tid & 31;
  const int wave = tid >> 5;
  const int g    = lane >> 4;     // half-wave (0/1)
  const int ln   = lane & 15;

  const int q0     = blockIdx.x * BLOCK_M;
  const int bh     = blockIdx.y;
  const int m_base = q0 + wave * 16;

  const size_t head = (size_t)bh * kS * kD;
  const float* qh = q + head;
  const float* kh = k + head;
  const float* vh = v + head;
  float*       oh = out + head;

  // ---- Q strip (16x64) -> two 16x32 bf16 A-fragments, fold scale*log2(e) ----
  const float qscale = 0.125f * 1.44269504088896f;  // D^-0.5 * log2(e)
  AFrag qa[2];
  {
    const float* qrow = qh + (size_t)(m_base + ln) * kD;
#pragma unroll
    for (int f = 0; f < 2; ++f)
#pragma unroll
      for (int vv = 0; vv < 8; ++vv) {
        const int k0 = 32 * f + 2 * vv + 16 * g;  // contiguous K order: k = 16g + s
        qa[f].u[vv] = bf16_2(qrow[k0] * qscale, qrow[k0 + 1] * qscale);
      }
  }

  // ones B-fragment (32x16 of bf16 1.0, reused for both K-chunks) for WMMA-based row sums
  AFrag ones;
#pragma unroll
  for (int vv = 0; vv < 8; ++vv) ones.u[vv] = 0x3F803F80u;

  const v8f vzero = {0.f, 0.f, 0.f, 0.f, 0.f, 0.f, 0.f, 0.f};
  v8f o[4] = {vzero, vzero, vzero, vzero};    // 16x64 output accumulator
  v8f lv   = vzero;                           // row-sum accumulator (replicated per col)
  float m_i[8];
#pragma unroll
  for (int r = 0; r < 8; ++r) m_i[r] = -1e30f;

  const int ntiles = (q0 + BLOCK_M) / BLOCK_N;   // causal bound for this block
  const int ldrow  = tid & 63;                   // K-load row
  const int ldseg  = tid >> 6;                   // K-load 16-float segment 0..3
  const int vkp    = (tid & 31) * 2;             // V-load key pair 0..62
  const int vseg   = tid >> 5;                   // V-load 8-float d segment 0..7

  for (int it = 0; it < ntiles; ++it) {
    const int n0 = it * BLOCK_N;
    __syncthreads();                             // previous tile fully consumed
    // ---- cooperative fp32 -> bf16 tile load: K row-major ----
    {
      const float4* ks = (const float4*)(kh + (size_t)(n0 + ldrow) * kD + ldseg * 16);
      unsigned* kd = (unsigned*)&k_lds[ldrow * LSTR + ldseg * 16];
      float4 a = ks[0], b = ks[1], c = ks[2], d = ks[3];
      kd[0] = bf16_2(a.x, a.y); kd[1] = bf16_2(a.z, a.w);
      kd[2] = bf16_2(b.x, b.y); kd[3] = bf16_2(b.z, b.w);
      kd[4] = bf16_2(c.x, c.y); kd[5] = bf16_2(c.z, c.w);
      kd[6] = bf16_2(d.x, d.y); kd[7] = bf16_2(d.z, d.w);
    }
    // ---- V transposed: thread owns key pair (vkp, vkp+1) x 8 d's -> packed u32 ----
    {
      const float* vp0 = vh + (size_t)(n0 + vkp) * kD + vseg * 8;
      const float4 a0 = ((const float4*)vp0)[0], a1 = ((const float4*)vp0)[1];
      const float4 b0 = ((const float4*)(vp0 + kD))[0], b1 = ((const float4*)(vp0 + kD))[1];
      const float r0[8] = {a0.x, a0.y, a0.z, a0.w, a1.x, a1.y, a1.z, a1.w};
      const float r1[8] = {b0.x, b0.y, b0.z, b0.w, b1.x, b1.y, b1.z, b1.w};
#pragma unroll
      for (int j = 0; j < 8; ++j)
        *(unsigned*)&vt_lds[(vseg * 8 + j) * LSTR + vkp] = bf16_2(r0[j], r1[j]);
    }
    __syncthreads();

    // ---- prefetch next tile's K/V lines into cache while WMMAs run ----
    if (it + 1 < ntiles) {
      __builtin_prefetch(kh + (size_t)(n0 + BLOCK_N + ldrow) * kD + ldseg * 16, 0, 3);
      __builtin_prefetch(vh + (size_t)(n0 + BLOCK_N + vkp) * kD + vseg * 8, 0, 3);
    }

    if (n0 > m_base + 15) continue;   // strip entirely above diagonal (wave-uniform)

    // ---- S = Q * K^T : 4 N-tiles x 2 K-chunks of v_wmma_f32_16x16x32_bf16 ----
    v8f s[4];
#pragma unroll
    for (int t = 0; t < 4; ++t) {
      const unsigned* krow = (const unsigned*)&k_lds[(16 * t + ln) * LSTR];
      AFrag b0, b1;
#pragma unroll
      for (int vv = 0; vv < 8; ++vv) {
        b0.u[vv] = krow[vv + 8 * g];        // d = 0..31   (B-layout: K=2v+16g)
        b1.u[vv] = krow[16 + vv + 8 * g];   // d = 32..63
      }
      v8f acc = vzero;
      acc = __builtin_amdgcn_wmma_f32_16x16x32_bf16(false, qa[0].v, false, b0.v,
                                                    (short)0, acc, false, false);
      acc = __builtin_amdgcn_wmma_f32_16x16x32_bf16(false, qa[1].v, false, b1.v,
                                                    (short)0, acc, false, false);
      s[t] = acc;
    }

    // ---- analytic causal mask on boundary tiles ----
    if (n0 + BLOCK_N - 1 > m_base) {
#pragma unroll
      for (int t = 0; t < 4; ++t) {
        const int col = n0 + 16 * t + ln;
#pragma unroll
        for (int r = 0; r < 8; ++r)
          if (col > m_base + r + 8 * g) s[t][r] = -1e30f;
      }
    }

    // ---- online softmax (log2 domain); row max via half-wave reduction ----
    float alpha[8];
#pragma unroll
    for (int r = 0; r < 8; ++r) {
      float mx = fmaxf(fmaxf(s[0][r], s[1][r]), fmaxf(s[2][r], s[3][r]));
#pragma unroll
      for (int off = 1; off < 16; off <<= 1)
        mx = fmaxf(mx, __shfl_xor(mx, off, 32));
      const float mnew = fmaxf(m_i[r], mx);
      alpha[r] = fast_exp2(m_i[r] - mnew);
      m_i[r]   = mnew;
    }
#pragma unroll
    for (int t = 0; t < 4; ++t)
#pragma unroll
      for (int r = 0; r < 8; ++r)
        s[t][r] = fast_exp2(s[t][r] - m_i[r]);
#pragma unroll
    for (int r = 0; r < 8; ++r) {
      lv[r] *= alpha[r];
#pragma unroll
      for (int t = 0; t < 4; ++t) o[t][r] *= alpha[r];
    }

    // ---- P: C-layout -> bf16 A-layout via per-wave LDS staging ----
    unsigned short* pst = &p_lds[wave * 16 * LSTR];
#pragma unroll
    for (int tp = 0; tp < 2; ++tp)
#pragma unroll
      for (int r = 0; r < 8; ++r) {
        const unsigned pr = bf16_2(s[2 * tp][r], s[2 * tp + 1][r]);
        unsigned short* base = &pst[(r + 8 * g) * LSTR + 32 * tp + ln];
        base[0]  = (unsigned short)(pr & 0xFFFFu);
        base[16] = (unsigned short)(pr >> 16);
      }

    AFrag pa[2];
#pragma unroll
    for (int f = 0; f < 2; ++f)
#pragma unroll
      for (int vv = 0; vv < 8; ++vv) {
        const int k0 = 32 * f + 2 * vv + 16 * g;
        pa[f].u[vv] = *(const unsigned*)&pst[ln * LSTR + k0];
      }

    // ---- row sums: lv += P * ones (replaces cross-lane sum reduction) ----
    lv = __builtin_amdgcn_wmma_f32_16x16x32_bf16(false, pa[0].v, false, ones.v,
                                                 (short)0, lv, false, false);
    lv = __builtin_amdgcn_wmma_f32_16x16x32_bf16(false, pa[1].v, false, ones.v,
                                                 (short)0, lv, false, false);

    // ---- O += P * V : 4 D-tiles x 2 K-chunks ----
#pragma unroll
    for (int t = 0; t < 4; ++t) {
      const unsigned* vrow = (const unsigned*)&vt_lds[(16 * t + ln) * LSTR];
      AFrag vb0, vb1;
#pragma unroll
      for (int vv = 0; vv < 8; ++vv) {
        vb0.u[vv] = vrow[vv + 8 * g];        // keys 0..31
        vb1.u[vv] = vrow[16 + vv + 8 * g];   // keys 32..63
      }
      o[t] = __builtin_amdgcn_wmma_f32_16x16x32_bf16(false, pa[0].v, false, vb0.v,
                                                     (short)0, o[t], false, false);
      o[t] = __builtin_amdgcn_wmma_f32_16x16x32_bf16(false, pa[1].v, false, vb1.v,
                                                     (short)0, o[t], false, false);
    }
  }

  // ---- epilogue: O / l ----
  float* orow = oh + (size_t)m_base * kD;
#pragma unroll
  for (int r = 0; r < 8; ++r) {
    const float inv = 1.f / lv[r];
    float* p = orow + (size_t)(r + 8 * g) * kD + ln;
    p[0]  = o[0][r] * inv;
    p[16] = o[1][r] * inv;
    p[32] = o[2][r] * inv;
    p[48] = o[3][r] * inv;
  }
}

extern "C" void kernel_launch(void* const* d_in, const int* in_sizes, int n_in,
                              void* d_out, int out_size, void* d_ws, size_t ws_size,
                              hipStream_t stream) {
  (void)in_sizes; (void)n_in; (void)d_ws; (void)ws_size; (void)out_size;
  const float* q = (const float*)d_in[0];
  const float* k = (const float*)d_in[1];
  const float* v = (const float*)d_in[2];
  // d_in[3] is the causal mask; it is tril by construction -> applied analytically.
  dim3 grid(kS / BLOCK_M, kB * kH);
  flash_attn_fwd<<<grid, 256, 0, stream>>>(q, k, v, (float*)d_out);
}
